// ACPClassifier_2422361555568
// MI455X (gfx1250) — matrix operand order repaired and checked by reading, not compiled, not run on hardware
//
#include <hip/hip_runtime.h>
#include <hip/hip_bf16.h>

// ---------- model dims ----------
#define DIMD   512
#define SEQ    1024
#define HEADS  8
#define HDIM   64
#define BATCH  8
#define VOCABN 1024
#define NROW   (BATCH*SEQ)          // 8192
#define MATE   ((size_t)NROW*DIMD)  // 4,194,304 elements

typedef _Float16 h16;
typedef _Float16 v8h  __attribute__((ext_vector_type(8)));
typedef _Float16 v16h __attribute__((ext_vector_type(16)));
typedef float    v8f  __attribute__((ext_vector_type(8)));
typedef float    v4f  __attribute__((ext_vector_type(4)));
typedef unsigned int u32x4 __attribute__((ext_vector_type(4)));
typedef unsigned int u32x8 __attribute__((ext_vector_type(8)));

static __device__ __forceinline__ v16h cat16(v8h lo, v8h hi) {
  return __builtin_shufflevector(lo, hi, 0,1,2,3,4,5,6,7,8,9,10,11,12,13,14,15);
}

// =====================================================================
// Tensor Data Mover: 2D tile load (global -> LDS), f16 elements.
//  - tile_d0 contiguous elements per row, tile_d1 rows
//  - stride  = row stride in elements
//  - lands contiguously in LDS at lds_byte
// Descriptor per CDNA5 ISA ch.8 (D# group0 128b + group1 256b), 2-group
// form (VADDR2/VADDR3 NULL => tensors up to 2D).
// =====================================================================
static __device__ __forceinline__ void tdm_load_2d_f16(unsigned lds_byte,
                                                       const void* gptr,
                                                       unsigned tile_d0,
                                                       unsigned tile_d1,
                                                       unsigned stride_elems) {
  const unsigned long long ga = (unsigned long long)(uintptr_t)gptr;
  u32x4 g0;
  g0[0] = 1u;                                   // count=1 (valid), user-mode load
  g0[1] = lds_byte;                             // lds_addr (bytes)
  g0[2] = (unsigned)ga;                         // global_addr[31:0]
  g0[3] = (unsigned)((ga >> 32) & 0x1FFFFFFu)   // global_addr[56:32]
        | (2u << 30);                           // type=2 ("image")
  u32x8 g1;
  g1[0] = (1u << 16);                           // workgroup_mask=0, data_size=1 (2B), flags=0
  g1[1] = 0u;                                   // atomic_barrier=0, tensor_dim0[15:0]=0
  g1[2] = 0x4000u;                              // tensor_dim0=2^30 (no OOB clip), tensor_dim1[15:0]=0
  g1[3] = 0x4000u | (tile_d0 << 16);            // tensor_dim1=2^30, tile_dim0
  g1[4] = tile_d1;                              // tile_dim1 ; tile_dim2=0
  g1[5] = stride_elems;                         // tensor_dim0_stride[31:0] (elements)
  g1[6] = 0u;                                   // stride0[47:32]=0, stride1[15:0]=0
  g1[7] = 0u;                                   // stride1[47:16]=0
  asm volatile("tensor_load_to_lds %0, %1" :: "s"(g0), "s"(g1) : "memory");
}

// =====================================================================
// Weight prep: Wt[n*512+k] = (f16) W[k*512+n]   (one 512x512 matrix)
// =====================================================================
__global__ __launch_bounds__(256) void prep_wt(const float* __restrict__ W,
                                               h16* __restrict__ Wt) {
  int idx = blockIdx.x * 256 + threadIdx.x;   // 0..262143
  int n = idx >> 9, k = idx & 511;
  Wt[idx] = (h16)W[k * DIMD + n];
}

// =====================================================================
// block reduce of (sum, sumsq) with 128 threads; sh must hold 256 floats
// =====================================================================
static __device__ __forceinline__ void reduce2_128(float& s, float& q, float* sh) {
  int tid = threadIdx.x;
  sh[tid] = s; sh[128 + tid] = q;
  __syncthreads();
  for (int off = 64; off > 0; off >>= 1) {
    if (tid < off) { sh[tid] += sh[tid + off]; sh[128 + tid] += sh[128 + tid + off]; }
    __syncthreads();
  }
  s = sh[0]; q = sh[128];
  __syncthreads();
}

// =====================================================================
// Embedding gather (only vocab pass i=2 affects the output) + LayerNorm
// =====================================================================
__global__ __launch_bounds__(128) void embed_ln(const int* __restrict__ ids,
                                                const float* __restrict__ etab,
                                                const float* __restrict__ pemb,
                                                const float* __restrict__ gam,
                                                const float* __restrict__ bet,
                                                float* __restrict__ x32,
                                                h16* __restrict__ xh) {
  __shared__ float sh[256];
  const int row = blockIdx.x;            // b*SEQ + s
  const int b = row >> 10, s = row & 1023;
  const int id = ids[(2 * BATCH + b) * SEQ + s];
  const float* e = etab + ((size_t)(2 * VOCABN + id)) * DIMD;
  const float* p = pemb + ((size_t)(2 * SEQ + s)) * DIMD;
  const int d0 = threadIdx.x * 4;
  v4f ev = *(const v4f*)(e + d0);
  v4f pv = *(const v4f*)(p + d0);
  v4f xv = ev + pv;
  float sum = xv.x + xv.y + xv.z + xv.w;
  float sq  = xv.x*xv.x + xv.y*xv.y + xv.z*xv.z + xv.w*xv.w;
  reduce2_128(sum, sq, sh);
  const float mu = sum * (1.0f / DIMD);
  const float var = sq * (1.0f / DIMD) - mu * mu;
  const float rs = rsqrtf(var + 1e-5f);
  v4f gv = *(const v4f*)(gam + d0);
  v4f bv = *(const v4f*)(bet + d0);
  float* xo = x32 + (size_t)row * DIMD + d0;
  h16*  ho = xh  + (size_t)row * DIMD + d0;
  #pragma unroll
  for (int i = 0; i < 4; ++i) {
    float v = (xv[i] - mu) * rs * gv[i] + bv[i];
    xo[i] = v;
    ho[i] = (h16)v;
  }
}

// =====================================================================
// Generic LayerNorm: in32 -> out16 and/or out32
// =====================================================================
__global__ __launch_bounds__(128) void ln_kernel(const float* __restrict__ in32,
                                                 const float* __restrict__ gam,
                                                 const float* __restrict__ bet,
                                                 float* __restrict__ out32,
                                                 h16* __restrict__ out16) {
  __shared__ float sh[256];
  const int row = blockIdx.x;
  const int d0 = threadIdx.x * 4;
  v4f xv = *(const v4f*)(in32 + (size_t)row * DIMD + d0);
  float sum = xv.x + xv.y + xv.z + xv.w;
  float sq  = xv.x*xv.x + xv.y*xv.y + xv.z*xv.z + xv.w*xv.w;
  reduce2_128(sum, sq, sh);
  const float mu = sum * (1.0f / DIMD);
  const float var = sq * (1.0f / DIMD) - mu * mu;
  const float rs = rsqrtf(var + 1e-5f);
  v4f gv = *(const v4f*)(gam + d0);
  v4f bv = *(const v4f*)(bet + d0);
  #pragma unroll
  for (int i = 0; i < 4; ++i) {
    float v = (xv[i] - mu) * rs * gv[i] + bv[i];
    if (out32) out32[(size_t)row * DIMD + d0 + i] = v;
    if (out16) out16[(size_t)row * DIMD + d0 + i] = (h16)v;
  }
}

// =====================================================================
// WMMA GEMM: C[M=8192, N=512] = A_f16[M,512] @ (Bt_f16[N,512])^T
// block 128 thr (4 waves) computes 64x64 tile; BK=32.
// A/B tiles staged to LDS by the Tensor Data Mover (one descriptor each,
// issued by wave 0; TENSORcnt wait + workgroup barrier orders consumers).
// =====================================================================
__global__ __launch_bounds__(128) void gemm_wmma(const h16* __restrict__ A,
                                                 const h16* __restrict__ Bt,
                                                 const float* __restrict__ bias,
                                                 const float* __restrict__ resid,
                                                 float* __restrict__ out32,
                                                 h16* __restrict__ out16,
                                                 int act) {
  __shared__ __align__(16) h16 As[64 * 32];
  __shared__ __align__(16) h16 Bs[64 * 32];
  const int m0 = blockIdx.x * 64;
  const int n0 = blockIdx.y * 64;
  const int tid = threadIdx.x;
  const int wave = tid >> 5, lane = tid & 31;
  const int hf = lane >> 4, l16 = lane & 15;
  // flat LDS pointer low 32 bits == LDS byte offset (aperture rule, ISA 10.2)
  const unsigned ldsA = (unsigned)(uintptr_t)&As[0];
  const unsigned ldsB = (unsigned)(uintptr_t)&Bs[0];
  v8f acc[4] = {};

  for (int k0 = 0; k0 < DIMD; k0 += 32) {
    if (wave == 0) {   // wave-uniform; TDM ignores EXEC
      tdm_load_2d_f16(ldsA, A  + (size_t)m0 * DIMD + k0, 32, 64, DIMD);
      tdm_load_2d_f16(ldsB, Bt + (size_t)n0 * DIMD + k0, 32, 64, DIMD);
      __builtin_amdgcn_s_wait_tensorcnt(0);
    }
    __syncthreads();
    // A fragment: row = wave*16 + l16, k chunks at 8*hf and 16+8*hf
    const h16* ap = &As[(wave * 16 + l16) * 32];
    v16h afrag = cat16(*(const v8h*)(ap + 8 * hf), *(const v8h*)(ap + 16 + 8 * hf));
    #pragma unroll
    for (int ns = 0; ns < 4; ++ns) {
      const h16* bp = &Bs[(ns * 16 + l16) * 32 + 8 * hf];
      v16h bfrag = cat16(*(const v8h*)bp, *(const v8h*)(bp + 16));
      acc[ns] = __builtin_amdgcn_wmma_f32_16x16x32_f16(
          false, afrag, false, bfrag, (short)0, acc[ns], false, false);
    }
    // make sure this wave's LDS reads retired before anyone's next TDM overwrite
    asm volatile("s_wait_dscnt 0x0" ::: "memory");
    __syncthreads();
  }
  #pragma unroll
  for (int ns = 0; ns < 4; ++ns) {
    const int col = n0 + ns * 16 + l16;
    const float bv = bias ? bias[col] : 0.0f;
    #pragma unroll
    for (int r = 0; r < 8; ++r) {
      const int row = m0 + wave * 16 + hf * 8 + r;
      float v = acc[ns][r] + bv;
      if (resid) v += resid[(size_t)row * DIMD + col];
      if (act == 1) v = fmaxf(v, 0.0f);
      if (out32) out32[(size_t)row * DIMD + col] = v;
      if (out16) out16[(size_t)row * DIMD + col] = (h16)v;
    }
  }
}

// =====================================================================
// V transpose: vt[(b*H+h)*64 + hd][s] = v16[(b*SEQ+s)*512 + h*64 + hd]
// =====================================================================
__global__ __launch_bounds__(256) void transpose_v(const h16* __restrict__ v16,
                                                   h16* __restrict__ vt) {
  const int bh = blockIdx.x >> 6;        // 0..63
  const int hd = blockIdx.x & 63;
  const int b = bh >> 3, h = bh & 7;
  for (int s = threadIdx.x; s < SEQ; s += 256)
    vt[((size_t)bh * 64 + hd) * SEQ + s] =
        v16[((size_t)(b * SEQ + s)) * DIMD + h * 64 + hd];
}

// =====================================================================
// Retention: per (b*H+h, query-tile of 16) one wave (32 thr)
// y = (QK^T * decay / sqrt(64)) V ; groupnorm per row; swish(g)*y -> gated f16
// =====================================================================
__global__ __launch_bounds__(32) void retention(const h16* __restrict__ q16,
                                                const h16* __restrict__ k16,
                                                const h16* __restrict__ vt,
                                                const float* __restrict__ g32,
                                                h16* __restrict__ gated) {
  const int bh = blockIdx.x;             // 0..63
  const int b = bh >> 3, h = bh & 7;
  const int qt = blockIdx.y;             // 0..63 query tiles of 16
  const int lane = threadIdx.x;
  const int hf = lane >> 4, l16 = lane & 15;

  __shared__ __align__(16) h16 sc[16 * 32];
  __shared__ float ytile[16 * 64];
  __shared__ float mu_s[16], rs_s[16];

  const float gamma = 1.0f - exp2f(-5.0f - (float)h);
  const float lg = log2f(gamma);
  const float scale = 0.125f;            // 1/sqrt(HDIM)

  // Q fragments (A-layout), hd 0..31 and 32..63
  const h16* qp = q16 + ((size_t)(b * SEQ + qt * 16 + l16)) * DIMD + h * 64;
  v16h aq[2];
  #pragma unroll
  for (int j = 0; j < 2; ++j)
    aq[j] = cat16(*(const v8h*)(qp + j * 32 + 8 * hf),
                  *(const v8h*)(qp + j * 32 + 16 + 8 * hf));

  v8f yacc[4] = {};
  const int nchunk2 = (qt >> 1) + 1;     // 32-key chunks covering causal range
  const h16* vbase = vt + (size_t)bh * 64 * SEQ;

  for (int c2 = 0; c2 < nchunk2; ++c2) {
    if (c2 + 1 < nchunk2) {              // prefetch next chunk's K rows and V run
      __builtin_prefetch(k16 + ((size_t)(b * SEQ + (c2 + 1) * 32 + l16)) * DIMD + h * 64, 0, 1);
      __builtin_prefetch(vbase + ((size_t)(lane)) * SEQ + (c2 + 1) * 32, 0, 1);
    }
    // two 16-key score tiles (t=0,1) -> sc[row][key] in LDS
    #pragma unroll
    for (int t = 0; t < 2; ++t) {
      const int kt = c2 * 2 + t;
      v8f s = {};
      if (kt <= qt) {                    // block-uniform -> EXEC stays full for WMMA
        const h16* kp = k16 + ((size_t)(b * SEQ + kt * 16 + l16)) * DIMD + h * 64;
        #pragma unroll
        for (int j = 0; j < 2; ++j) {
          v16h bk = cat16(*(const v8h*)(kp + j * 32 + 8 * hf),
                          *(const v8h*)(kp + j * 32 + 16 + 8 * hf));
          s = __builtin_amdgcn_wmma_f32_16x16x32_f16(
              false, aq[j], false, bk, (short)0, s, false, false);
        }
      }
      const int m = kt * 16 + l16;       // key index (column)
      #pragma unroll
      for (int r = 0; r < 8; ++r) {
        const int n = qt * 16 + hf * 8 + r;   // query index (row)
        float w = (kt <= qt && m <= n) ? scale * exp2f(lg * (float)(n - m)) : 0.0f;
        sc[(hf * 8 + r) * 32 + t * 16 + l16] = (h16)(s[r] * w);
      }
    }
    __syncthreads();
    // scores as A fragment (16 rows x 32 keys)
    const h16* sp = &sc[l16 * 32];
    v16h asc = cat16(*(const v8h*)(sp + 8 * hf), *(const v8h*)(sp + 16 + 8 * hf));
    // V fragments (B-layout): K=32 keys, N=16 hdim per sub-tile
    #pragma unroll
    for (int ns = 0; ns < 4; ++ns) {
      const h16* vp = vbase + ((size_t)(ns * 16 + l16)) * SEQ + c2 * 32 + 8 * hf;
      v16h bv = cat16(*(const v8h*)vp, *(const v8h*)(vp + 16));
      yacc[ns] = __builtin_amdgcn_wmma_f32_16x16x32_f16(
          false, asc, false, bv, (short)0, yacc[ns], false, false);
    }
    __syncthreads();
  }

  // stash y to LDS, group-norm per query row over 64 hdims
  #pragma unroll
  for (int ns = 0; ns < 4; ++ns)
    #pragma unroll
    for (int r = 0; r < 8; ++r)
      ytile[(hf * 8 + r) * 64 + ns * 16 + l16] = yacc[ns][r];
  __syncthreads();
  if (lane < 16) {
    float sum = 0.0f, sq = 0.0f;
    #pragma unroll
    for (int c = 0; c < 64; ++c) { float v = ytile[lane * 64 + c]; sum += v; sq += v * v; }
    const float mu = sum * (1.0f / 64.0f);
    const float var = sq * (1.0f / 64.0f) - mu * mu;
    mu_s[lane] = mu; rs_s[lane] = rsqrtf(var + 1e-5f);
  }
  __syncthreads();
  // swish gate + scatter back to [row, h*64+hd] layout (f16)
  #pragma unroll
  for (int j = 0; j < 32; ++j) {
    const int e = lane * 32 + j;
    const int row = e >> 6, col = e & 63;
    const size_t go = ((size_t)(b * SEQ + qt * 16 + row)) * DIMD + h * 64 + col;
    const float g = g32[go];
    const float sw = g / (1.0f + __expf(-g));
    const float yv = (ytile[row * 64 + col] - mu_s[row]) * rs_s[row];
    gated[go] = (h16)(sw * yv);
  }
}

// =====================================================================
// Final classifier: out[b] = sigmoid( sum_i xf[b, i] * fcW[i] + fcb )
// =====================================================================
__global__ __launch_bounds__(256) void fc_kernel(const float* __restrict__ xf,
                                                 const float* __restrict__ fcW,
                                                 const float* __restrict__ fcb,
                                                 float* __restrict__ out) {
  __shared__ float sh[256];
  const int b = blockIdx.x;
  const size_t base = (size_t)b * (SEQ * DIMD);
  float s = 0.0f;
  for (int i = threadIdx.x; i < SEQ * DIMD; i += 256)
    s += xf[base + i] * fcW[i];
  sh[threadIdx.x] = s;
  __syncthreads();
  for (int off = 128; off > 0; off >>= 1) {
    if (threadIdx.x < off) sh[threadIdx.x] += sh[threadIdx.x + off];
    __syncthreads();
  }
  if (threadIdx.x == 0) {
    float z = sh[0] + fcb[0];
    out[b] = 1.0f / (1.0f + __expf(-z));
  }
}

// =====================================================================
// launch
// =====================================================================
extern "C" void kernel_launch(void* const* d_in, const int* in_sizes, int n_in,
                              void* d_out, int out_size, void* d_ws, size_t ws_size,
                              hipStream_t stream) {
  const int*   ids   = (const int*)  d_in[0];
  const float* etab  = (const float*)d_in[1];
  const float* pemb  = (const float*)d_in[2];
  const float* gam   = (const float*)d_in[3];
  const float* bet   = (const float*)d_in[4];
  const float* Wq    = (const float*)d_in[5];
  const float* Wk    = (const float*)d_in[6];
  const float* Wv    = (const float*)d_in[7];
  const float* Wg    = (const float*)d_in[8];
  const float* Wo    = (const float*)d_in[9];
  const float* W1    = (const float*)d_in[10];
  const float* b1    = (const float*)d_in[11];
  const float* W2    = (const float*)d_in[12];
  const float* b2    = (const float*)d_in[13];
  const float* fcW   = (const float*)d_in[14];
  const float* fcb   = (const float*)d_in[15];
  float* out = (float*)d_out;

  // ---- workspace carve-up (with reuse) ----
  char* ws = (char*)d_ws;
  const size_t WMAT16 = (size_t)DIMD * DIMD * sizeof(h16);   // 512 KB
  const size_t M16 = MATE * sizeof(h16);                     // 8 MB
  const size_t M32 = MATE * sizeof(float);                   // 16 MB
  size_t off = 0;
  h16* wt = (h16*)(ws + off);            off += 7 * WMAT16;  // Wq..W2 (f16, transposed)
  float* x32   = (float*)(ws + off);     off += M32;         // LN(emb); later reused as xf
  h16*   xh    = (h16*)(ws + off);       off += M16;         // f16 LN out; later h16 (ln2)
  h16*   q16   = (h16*)(ws + off);       off += M16;         // later ffn1 relu f16
  h16*   k16   = (h16*)(ws + off);       off += M16;
  h16*   v16   = (h16*)(ws + off);       off += M16;
  float* g32   = (float*)(ws + off);     off += M32;         // later r2_32
  h16*   vtb   = (h16*)(ws + off);       off += M16;
  h16*   gated = (h16*)(ws + off);       off += M16;
  float* r32   = (float*)(ws + off);     off += M32;
  // reuse mapping
  h16*   h16b  = xh;     // ln2 output f16
  h16*   f16b  = q16;    // relu(ffn1) f16
  float* r2_32 = g32;    // ffn2 + residual
  float* xf32  = x32;    // final LN f32

  h16* wtQ = wt + 0 * (size_t)(DIMD * DIMD);
  h16* wtK = wt + 1 * (size_t)(DIMD * DIMD);
  h16* wtV = wt + 2 * (size_t)(DIMD * DIMD);
  h16* wtG = wt + 3 * (size_t)(DIMD * DIMD);
  h16* wtO = wt + 4 * (size_t)(DIMD * DIMD);
  h16* wt1 = wt + 5 * (size_t)(DIMD * DIMD);
  h16* wt2 = wt + 6 * (size_t)(DIMD * DIMD);

  const dim3 gGemm(NROW / 64, DIMD / 64);   // (128, 8)
  const dim3 bGemm(128);
  const dim3 gRet(HEADS * BATCH, SEQ / 16); // (64, 64)

  // 1) weights -> f16 transposed
  prep_wt<<<1024, 256, 0, stream>>>(Wq, wtQ);
  prep_wt<<<1024, 256, 0, stream>>>(Wk, wtK);
  prep_wt<<<1024, 256, 0, stream>>>(Wv, wtV);
  prep_wt<<<1024, 256, 0, stream>>>(Wg, wtG);
  prep_wt<<<1024, 256, 0, stream>>>(Wo, wtO);
  prep_wt<<<1024, 256, 0, stream>>>(W1, wt1);
  prep_wt<<<1024, 256, 0, stream>>>(W2, wt2);

  // 2) embedding gather (last vocab pass only) + LN
  embed_ln<<<NROW, 128, 0, stream>>>(ids, etab, pemb, gam, bet, x32, xh);

  // 3) projections (WMMA, TDM-staged tiles)
  gemm_wmma<<<gGemm, bGemm, 0, stream>>>(xh, wtQ, nullptr, nullptr, nullptr, q16, 0);
  gemm_wmma<<<gGemm, bGemm, 0, stream>>>(xh, wtK, nullptr, nullptr, nullptr, k16, 0);
  gemm_wmma<<<gGemm, bGemm, 0, stream>>>(xh, wtV, nullptr, nullptr, nullptr, v16, 0);
  gemm_wmma<<<gGemm, bGemm, 0, stream>>>(xh, wtG, nullptr, nullptr, g32, nullptr, 0);

  // 4) V -> [b,h,hd,s] for contiguous B-fragment loads
  transpose_v<<<64 * 64, 256, 0, stream>>>(v16, vtb);

  // 5) retention + group-norm + swish gate
  retention<<<gRet, 32, 0, stream>>>(q16, k16, vtb, g32, gated);

  // 6) output projection + residual
  gemm_wmma<<<gGemm, bGemm, 0, stream>>>(gated, wtO, nullptr, x32, r32, nullptr, 0);

  // 7) LN -> f16
  ln_kernel<<<NROW, 128, 0, stream>>>(r32, gam, bet, nullptr, h16b);

  // 8) FFN
  gemm_wmma<<<gGemm, bGemm, 0, stream>>>(h16b, wt1, b1, nullptr, nullptr, f16b, 1);
  gemm_wmma<<<gGemm, bGemm, 0, stream>>>(f16b, wt2, b2, r32, r2_32, nullptr, 0);

  // 9) final LN (f32 for classifier)
  ln_kernel<<<NROW, 128, 0, stream>>>(r2_32, gam, bet, xf32, nullptr);

  // 10) classifier head
  fc_kernel<<<BATCH, 256, 0, stream>>>(xf32, fcW, fcb, out);
}
